// VectorQuantizer_1460288881296
// MI455X (gfx1250) — matrix operand order repaired and checked by reading, not compiled, not run on hardware
//
#include <hip/hip_runtime.h>
#include <hip/hip_bf16.h>

// ---------------------------------------------------------------------------
// VQ (vector quantizer): z [B,D] f32, codebook [K,D] f32
// out = [ z_q_st (B*D f32) | vq_loss (1 f32) | indices (B f32) ]
// Strategy: bf16 WMMA GEMM (z . e^T) fused with running argmin over
// score = |e|^2 - 2 z.e  (|z|^2 constant per row), codebook tiles DMA'd
// into LDS by the Tensor Data Mover, then exact f32 gather + loss.
// ---------------------------------------------------------------------------

#define VQ_B 32768
#define VQ_D 512
#define VQ_K 8192

typedef __bf16 bf16_t;
typedef __attribute__((ext_vector_type(16))) __bf16 v16bf;
typedef __attribute__((ext_vector_type(8)))  __bf16 v8bf;
typedef __attribute__((ext_vector_type(4)))  __bf16 v4bf;
typedef __attribute__((ext_vector_type(8)))  float  v8f;
typedef __attribute__((ext_vector_type(4)))  unsigned int u32x4;
typedef __attribute__((ext_vector_type(8)))  int  i32x8;
typedef __attribute__((ext_vector_type(4)))  int  i32x4;

#if __has_builtin(__builtin_amdgcn_tensor_load_to_lds)
#define VQ_TDM 1
#else
#define VQ_TDM 0
#endif

#define CAT16(lo, hi) __builtin_shufflevector((lo), (hi), 0,1,2,3,4,5,6,7,8,9,10,11,12,13,14,15)

__device__ __forceinline__ v8f wmma_bf16(v16bf a, v16bf b, v8f c) {
  // D = A(16x32 bf16) * B(32x16 bf16) + C(16x16 f32)
  return __builtin_amdgcn_wmma_f32_16x16x32_bf16(
      /*neg_a=*/false, a, /*neg_b=*/false, b,
      /*c_mod=*/(short)0, c, /*reuse_a=*/false, /*reuse_b=*/false);
}

// ---------------------------------------------------------------------------
// Prep 1: convert z (f32 -> bf16), flat
// ---------------------------------------------------------------------------
__global__ void vq_prep_z(const float* __restrict__ z, bf16_t* __restrict__ zb) {
  size_t i = (size_t)blockIdx.x * blockDim.x + threadIdx.x;  // i < B*D/4 exactly
  float4 v = ((const float4*)z)[i];
  v4bf o = {(__bf16)v.x, (__bf16)v.y, (__bf16)v.z, (__bf16)v.w};
  ((v4bf*)zb)[i] = o;
}

// ---------------------------------------------------------------------------
// Prep 2: codebook f32 -> bf16 + e_sq[K]; also zero the loss slot in out
// One wave (32 lanes) per codebook row; 8 waves per block.
// ---------------------------------------------------------------------------
__global__ void vq_prep_cb(const float* __restrict__ cb, bf16_t* __restrict__ cbb,
                           float* __restrict__ esq, float* __restrict__ out) {
  int wave = threadIdx.x >> 5;
  int lane = threadIdx.x & 31;
  int row = blockIdx.x * 8 + wave;

  const float4* src = (const float4*)cb + (size_t)row * (VQ_D / 4) + lane * 4;
  v4bf*         dst = (v4bf*)cbb        + (size_t)row * (VQ_D / 4) + lane * 4;

  float s = 0.f;
#pragma unroll
  for (int j = 0; j < 4; ++j) {
    float4 v = src[j];
    v4bf o = {(__bf16)v.x, (__bf16)v.y, (__bf16)v.z, (__bf16)v.w};
    dst[j] = o;
    s += v.x * v.x + v.y * v.y + v.z * v.z + v.w * v.w;
  }
#pragma unroll
  for (int m = 16; m >= 1; m >>= 1) s += __shfl_xor(s, m, 32);
  if (lane == 0) esq[row] = s;

  if (blockIdx.x == 0 && threadIdx.x == 0)
    out[(size_t)VQ_B * VQ_D] = 0.f;  // loss accumulator slot, re-zeroed per launch
}

// ---------------------------------------------------------------------------
// Main fused kernel. Block = 256 threads = 8 waves; 128 z-rows per block.
// ---------------------------------------------------------------------------
#define PITCH 520                 // 512 + 8 bf16 pad (TDM pad: 16B per 1024B row)
#define NTILES (VQ_K / 16)        // 512 codebook tiles of 16 entries

#if VQ_TDM
// TDM: DMA one 16x512 bf16 tile (row-major, stride 512 elems) from global
// into LDS with 16B padding after each 1024B row (-> PITCH=520 elems).
__device__ __forceinline__ void tdm_load_tile(const bf16_t* gsrc, void* lds_dst) {
  unsigned long long va = (unsigned long long)(size_t)gsrc;
  unsigned int ldsb = (unsigned int)(size_t)lds_dst;  // flat LDS addr: low 32 bits = LDS offset

  u32x4 g0;
  g0.x = 1u;                                       // count=1 (valid user descriptor)
  g0.y = ldsb;                                     // lds_addr (bytes)
  g0.z = (unsigned int)va;                         // global_addr[31:0]
  g0.w = (unsigned int)(va >> 32) | 0x80000000u;   // global_addr[56:32] | type=2 (bits 127:126)

  i32x8 g1;
  g1[0] = (1 << 16)    // data_size = 2 bytes
        | (1 << 20)    // pad_enable
        | (7 << 22)    // pad_interval: 256 DWORDs = 1024B
        | (3 << 25);   // pad_amount:   4 DWORDs  = 16B
  g1[1] = (int)(512u << 16);   // tensor_dim0 = 512 (bits 79:48, low part)
  g1[2] = (int)(16u << 16);    // tensor_dim0 hi = 0 | tensor_dim1 = 16 (low part)
  g1[3] = (int)(512u << 16);   // tensor_dim1 hi = 0 | tile_dim0 = 512
  g1[4] = 16;                  // tile_dim1 = 16, tile_dim2 = 0 (2D)
  g1[5] = 512;                 // tensor_dim0_stride = 512 elems
  g1[6] = 0;
  g1[7] = 0;

  i32x4 gz = {0, 0, 0, 0};
#if __clang_major__ >= 23
  i32x8 gz8 = {0, 0, 0, 0, 0, 0, 0, 0};
  __builtin_amdgcn_tensor_load_to_lds(g0, g1, gz, gz, gz8, 0);
#else
  __builtin_amdgcn_tensor_load_to_lds(g0, g1, gz, gz, 0);
#endif
}
#endif

__global__ __launch_bounds__(256)
void vq_main(const bf16_t* __restrict__ zb, const bf16_t* __restrict__ cbb,
             const float* __restrict__ esq, const float* __restrict__ z,
             const float* __restrict__ cb, float* __restrict__ out) {
  __shared__ __align__(16) __bf16 s_tile[2 * 16 * PITCH];
  __shared__ int s_idx[128];

  const int tid  = threadIdx.x;
  const int wave = tid >> 5;
  const int lane = tid & 31;
  const int l16  = lane & 15;
  const int h    = lane >> 4;          // half-lane selector
  const int rowBase = blockIdx.x * 128;
  const int grow = rowBase + wave * 16 + l16;   // this lane's z row

  // ---- Load wave's A strip (16 rows x 512 K, bf16) into registers,
  //      honoring the 16-bit A-matrix VGPR layout (half0: K0-7,16-23; half1: K8-15,24-31)
  v16bf afrag[16];
  {
    const v8bf* zrow = (const v8bf*)(zb + (size_t)grow * VQ_D);
#pragma unroll
    for (int c = 0; c < 16; ++c) {
      v8bf lo = zrow[c * 4 + h];        // elems c*32 + h*8
      v8bf hi = zrow[c * 4 + 2 + h];    // elems c*32 + 16 + h*8
      afrag[c] = CAT16(lo, hi);
    }
  }

  // running argmin state: 8 M-rows per lane (C-tile layout)
  float minv[8];
  int   mini[8];
#pragma unroll
  for (int m = 0; m < 8; ++m) { minv[m] = 3.402823466e38f; mini[m] = 0; }

#if !VQ_TDM
  const int ldRow = tid >> 4;     // fallback loader: 0..15
  const int ldSeg = tid & 15;     // 0..15, 64B each
  const uint4* gsrc = (const uint4*)cbb;
#endif

  // ---- prologue: tile 0 -> buffer 0
#if VQ_TDM
  if (wave == 0) {
    tdm_load_tile(cbb, s_tile);
    __builtin_amdgcn_s_wait_tensorcnt(0);
  }
#else
  {
    size_t gb = (size_t)ldRow * (VQ_D / 8) + ldSeg * 4;
    uint4 p0 = gsrc[gb], p1 = gsrc[gb + 1], p2 = gsrc[gb + 2], p3 = gsrc[gb + 3];
    uint4* ld = (uint4*)s_tile;
    size_t lb = (size_t)ldRow * (PITCH / 8) + ldSeg * 4;
    ld[lb] = p0; ld[lb + 1] = p1; ld[lb + 2] = p2; ld[lb + 3] = p3;
  }
#endif
  __syncthreads();

  for (int t = 0; t < NTILES; ++t) {
    const bool hasNext = (t + 1 < NTILES);

    // ---- kick off async DMA of next tile into the other buffer
#if VQ_TDM
    if (hasNext && wave == 0)
      tdm_load_tile(cbb + (size_t)(t + 1) * 16 * VQ_D,
                    s_tile + (size_t)((t + 1) & 1) * (16 * PITCH));
#else
    uint4 p0, p1, p2, p3;
    if (hasNext) {
      size_t gb = (size_t)(t + 1) * 16 * (VQ_D / 8) + (size_t)ldRow * (VQ_D / 8) + ldSeg * 4;
      p0 = gsrc[gb]; p1 = gsrc[gb + 1]; p2 = gsrc[gb + 2]; p3 = gsrc[gb + 3];
    }
#endif

    // ---- compute on current buffer: 16 WMMAs, two accumulator chains
    const __bf16* buf = s_tile + (size_t)(t & 1) * (16 * PITCH);
    const v8bf* brow = (const v8bf*)(buf + (size_t)l16 * PITCH);
    v8f acc0 = {0,0,0,0,0,0,0,0};
    v8f acc1 = {0,0,0,0,0,0,0,0};
#pragma unroll
    for (int c = 0; c < 16; c += 2) {
      // B layout: half0 K0-7,16-23; half1 K8-15,24-31 (matches A)
      v8bf bl0 = brow[c * 4 + h];
      v8bf bh0 = brow[c * 4 + 2 + h];
      v8bf bl1 = brow[(c + 1) * 4 + h];
      v8bf bh1 = brow[(c + 1) * 4 + 2 + h];
      acc0 = wmma_bf16(afrag[c],     CAT16(bl0, bh0), acc0);
      acc1 = wmma_bf16(afrag[c + 1], CAT16(bl1, bh1), acc1);
    }
    v8f accs = acc0 + acc1;

    // score = e_sq[n] - 2 * (z . e_n)
    const int ncol = t * 16 + l16;
    const float ev = esq[ncol];
#pragma unroll
    for (int m = 0; m < 8; ++m) {
      float s = ev - 2.0f * accs[m];
      if (s < minv[m]) { minv[m] = s; mini[m] = ncol; }
    }

    // ---- make next tile visible before the barrier
#if VQ_TDM
    if (hasNext && wave == 0)
      __builtin_amdgcn_s_wait_tensorcnt(0);
#else
    if (hasNext) {
      uint4* ld = (uint4*)(s_tile + (size_t)((t + 1) & 1) * (16 * PITCH));
      size_t lb = (size_t)ldRow * (PITCH / 8) + ldSeg * 4;
      ld[lb] = p0; ld[lb + 1] = p1; ld[lb + 2] = p2; ld[lb + 3] = p3;
    }
#endif
    __syncthreads();
  }

  // ---- cross-lane argmin butterfly within each 16-lane half (N reduction)
#pragma unroll
  for (int m = 0; m < 8; ++m) {
#pragma unroll
    for (int mask = 1; mask < 16; mask <<= 1) {
      float ov = __shfl_xor(minv[m], mask, 32);
      int   oi = __shfl_xor(mini[m], mask, 32);
      if (ov < minv[m] || (ov == minv[m] && oi < mini[m])) { minv[m] = ov; mini[m] = oi; }
    }
  }
  // lane 0 holds rows 0-7, lane 16 holds rows 8-15 of the wave strip
  if (l16 == 0) {
    int base = wave * 16 + h * 8;
#pragma unroll
    for (int m = 0; m < 8; ++m) {
      s_idx[base + m] = mini[m];
      out[(size_t)VQ_B * VQ_D + 1 + (rowBase + base + m)] = (float)mini[m];
    }
  }
  __syncthreads();

  // ---- exact f32 gather of z_q + straight-through output + loss partials
  {
    int r = tid >> 1;              // 0..127
    int hs = tid & 1;              // which half of the 512-dim row
    int gr = rowBase + r;
    int idx = s_idx[r];
    const float4* qsrc = (const float4*)(cb + (size_t)idx * VQ_D) + hs * 64;
    const float4* zsrc = (const float4*)(z  + (size_t)gr  * VQ_D) + hs * 64;
    float4*       dstp = (float4*)(out + (size_t)gr * VQ_D) + hs * 64;
    float part = 0.f;
#pragma unroll 4
    for (int j = 0; j < 64; ++j) {
      float4 q = qsrc[j];
      float4 zz = zsrc[j];
      dstp[j] = q;   // z_q_st == z_q numerically
      float dx = q.x - zz.x, dy = q.y - zz.y, dz = q.z - zz.z, dw = q.w - zz.w;
      part += dx * dx + dy * dy + dz * dz + dw * dw;
    }
    // vq_loss = (1 + 0.25) * mean((z_q - z)^2)
    atomicAdd(out + (size_t)VQ_B * VQ_D, part * (1.25f / ((float)VQ_B * (float)VQ_D)));
  }
}

// ---------------------------------------------------------------------------
extern "C" void kernel_launch(void* const* d_in, const int* in_sizes, int n_in,
                              void* d_out, int out_size, void* d_ws, size_t ws_size,
                              hipStream_t stream) {
  const float* z  = (const float*)d_in[0];
  const float* cb = (const float*)d_in[1];
  float* out = (float*)d_out;

  bf16_t* zb  = (bf16_t*)d_ws;
  bf16_t* cbb = (bf16_t*)((char*)d_ws + (size_t)VQ_B * VQ_D * 2);
  float*  esq = (float*)((char*)d_ws + (size_t)VQ_B * VQ_D * 2 + (size_t)VQ_K * VQ_D * 2);

  vq_prep_z<<<(VQ_B * VQ_D) / 1024, 256, 0, stream>>>(z, zb);
  vq_prep_cb<<<VQ_K / 8, 256, 0, stream>>>(cb, cbb, esq, out);
  vq_main<<<VQ_B / 128, 256, 0, stream>>>(zb, cbb, esq, z, cb, out);
}
